// MambaSequenceModel_34239479284056
// MI455X (gfx1250) — compile-verified, matrix operand order checked by reading, not run on hardware
//
#include <hip/hip_runtime.h>
#include <hip/hip_bf16.h>

// ---------------- model dims (compile-time) ----------------
#define BATCH     32
#define SEQ       512
#define MTOK      (BATCH * SEQ)      // 16384
#define IN_DIM    16
#define NCLS      2
#define DMODEL    128
#define NLAYER    4
#define DSTATE    16
#define DINNER    256
#define DTRANK    8
#define DCONV     4
#define XPROJ_N   (DTRANK + 2 * DSTATE) // 40

typedef __attribute__((ext_vector_type(16))) _Float16 v16h;
typedef __attribute__((ext_vector_type(8)))  _Float16 v8h;
typedef __attribute__((ext_vector_type(8)))  float    v8f;
typedef __attribute__((ext_vector_type(4)))  int      v4i;

__device__ inline v8h zero_v8h() {
  v8h z;
#pragma unroll
  for (int i = 0; i < 8; ++i) z[i] = (_Float16)0;
  return z;
}
__device__ inline v8f zero_v8f() {
  v8f z;
#pragma unroll
  for (int i = 0; i < 8; ++i) z[i] = 0.0f;
  return z;
}

// ---- CDNA5 async global->LDS (16B per lane), tracked by ASYNCcnt ----
__device__ __forceinline__ void async_b128(const _Float16* g, _Float16* l) {
#if __has_builtin(__builtin_amdgcn_global_load_async_to_lds_b128)
  __builtin_amdgcn_global_load_async_to_lds_b128(
      (__attribute__((address_space(1))) v4i*)g,
      (__attribute__((address_space(3))) v4i*)l, 0, 0);
#else
  unsigned loff = (unsigned)(uintptr_t)(__attribute__((address_space(3))) void*)l;
  asm volatile("global_load_async_to_lds_b128 %0, %1, off"
               :: "v"(loff), "v"(g) : "memory");
#endif
}

__device__ __forceinline__ void wait_async0() {
#if __has_builtin(__builtin_amdgcn_s_wait_asynccnt)
  __builtin_amdgcn_s_wait_asynccnt(0);
#else
  asm volatile("s_wait_asynccnt 0x0" ::: "memory");
#endif
}

// ---------------- f32 -> f16 cast ----------------
__global__ void k_cast_f16(const float* __restrict__ src, _Float16* __restrict__ dst, int n) {
  int i = blockIdx.x * blockDim.x + threadIdx.x;
  if (i < n) dst[i] = (_Float16)src[i];
}

// ---------------- input projection: h = x @ w_in^T + b_in ----------------
__global__ void k_in_proj(const float* __restrict__ x, const float* __restrict__ w_in,
                          const float* __restrict__ b_in, float* __restrict__ h) {
  int idx = blockIdx.x * blockDim.x + threadIdx.x;   // [0, MTOK*DMODEL)
  int d = idx & (DMODEL - 1);
  int m = idx >> 7;
  float acc = b_in[d];
  const float* xr = x + (size_t)m * IN_DIM;
  const float* wr = w_in + (size_t)d * IN_DIM;
#pragma unroll
  for (int k = 0; k < IN_DIM; ++k) acc += xr[k] * wr[k];
  h[idx] = acc;
}

// ---------------- rmsnorm -> f16 ----------------
__global__ void k_rmsnorm_f16(const float* __restrict__ h, const float* __restrict__ w,
                              _Float16* __restrict__ out) {
  int m = blockIdx.x;
  int t = threadIdx.x;             // 128 threads
  float v = h[(size_t)m * DMODEL + t];
  __shared__ float red[DMODEL];
  red[t] = v * v;
  __syncthreads();
#pragma unroll
  for (int s = DMODEL / 2; s > 0; s >>= 1) {
    if (t < s) red[t] += red[t + s];
    __syncthreads();
  }
  float inv = rsqrtf(red[0] / (float)DMODEL + 1e-5f);
  out[(size_t)m * DMODEL + t] = (_Float16)(v * inv * w[t]);
}

// ---------------- WMMA GEMM: C[M,NTOT] = A[M,K](f16) * W[NTOT,K]^T(f16) (+ resid) ----
// block = 256 threads (8 waves), block tile 128x64, wave tile 16x64.
// Tiles staged with GLOBAL_LOAD_ASYNC_TO_LDS_B128 (ASYNCcnt), fragments via ds_load_b128.
template <int K, int NTOT, bool ADD_RESID>
__global__ __launch_bounds__(256)
void wmma_gemm(const _Float16* __restrict__ A, const _Float16* __restrict__ W,
               float* __restrict__ Cout, const float* __restrict__ resid) {
  __shared__ _Float16 As[128 * 40];   // 128 rows, stride 40 halves (80B, 16B aligned)
  __shared__ _Float16 Bs[64 * 40];    // stored [N][K] so B-fragment mirrors A pattern

  const int tid  = threadIdx.x;
  const int wave = tid >> 5;
  const int lane = tid & 31;
  const int m_blk = blockIdx.x * 128;
  const int n_blk = blockIdx.y * 64;

  // zero the padded B rows once (async loads below are predicated n < NTOT)
  if ((NTOT & 63) != 0) {
    int r  = tid >> 2;
    int cc = (tid & 3) * 8;
    if (n_blk + r >= NTOT) *(v8h*)(Bs + r * 40 + cc) = zero_v8h();
  }

  v8f acc[4];
#pragma unroll
  for (int i = 0; i < 4; ++i) acc[i] = zero_v8f();

  for (int kk = 0; kk < K; kk += 32) {
    // --- stage A tile (async): 128 x 32 halves, 2 x 16B per thread ---
    {
      int r  = tid >> 1;
      int cc = (tid & 1) * 16;
      const _Float16* src = A + (size_t)(m_blk + r) * K + kk + cc;
      _Float16* dst = As + r * 40 + cc;
      async_b128(src, dst);
      async_b128(src + 8, dst + 8);
      if (kk + 32 < K) __builtin_prefetch(src + 32, 0, 1);   // next K tile -> cache
    }
    // --- stage B tile (async): 64 x 32 halves from W[N,K], 16B per thread ---
    {
      int r  = tid >> 2;
      int cc = (tid & 3) * 8;
      int n  = n_blk + r;
      if (n < NTOT)
        async_b128(W + (size_t)n * K + kk + cc, Bs + r * 40 + cc);
    }
    wait_async0();       // drain this wave's async copies
    __syncthreads();     // make every wave's LDS writes visible

    // --- fragments + WMMA (manually unrolled: reuse_a flag must be an ICE) ---
    const int klo = (lane & 16) ? 8 : 0;            // lanes16-31 hold K=8..15 / 24..31
    union Frag { v16h v; v8h h[2]; };
    Frag af;
    const _Float16* ra = As + (wave * 16 + (lane & 15)) * 40;
    af.h[0] = *(const v8h*)(ra + klo);
    af.h[1] = *(const v8h*)(ra + klo + 16);

#define WMMA_STEP(NT, RA)                                                     \
    {                                                                         \
      Frag bf;                                                                \
      const _Float16* rb = Bs + ((NT) * 16 + (lane & 15)) * 40;               \
      bf.h[0] = *(const v8h*)(rb + klo);                                      \
      bf.h[1] = *(const v8h*)(rb + klo + 16);                                 \
      acc[NT] = __builtin_amdgcn_wmma_f32_16x16x32_f16(                       \
          false, af.v, false, bf.v, (short)0, acc[NT], RA, false);            \
    }
    WMMA_STEP(0, false)
    WMMA_STEP(1, true)    // identical to previous op, same A fragment
    WMMA_STEP(2, true)
    WMMA_STEP(3, false)
#undef WMMA_STEP
    __syncthreads();
  }

  // --- store: VGPR i -> M = i (+8 for lanes 16-31), N = lane%16 ---
  const int mrow  = m_blk + wave * 16 + ((lane & 16) ? 8 : 0);
  const int ncol0 = n_blk + (lane & 15);
#pragma unroll
  for (int nt = 0; nt < 4; ++nt) {
    int n = ncol0 + nt * 16;
    if ((NTOT & 63) != 0 && n >= NTOT) continue;
#pragma unroll
    for (int i = 0; i < 8; ++i) {
      size_t idx = (size_t)(mrow + i) * NTOT + n;
      float v = acc[nt][i];
      if (ADD_RESID) v += resid[idx];
      Cout[idx] = v;
    }
  }
}

// ---------------- causal depthwise conv (d_conv=4) + SiLU ----------------
__global__ void k_conv_silu(const float* __restrict__ xz, const float* __restrict__ cw,
                            const float* __restrict__ cb, float* __restrict__ u,
                            _Float16* __restrict__ u_h) {
  int idx = blockIdx.x * blockDim.x + threadIdx.x;    // [0, MTOK*DINNER)
  int c  = idx & (DINNER - 1);
  int ml = idx >> 8;
  int l  = ml & (SEQ - 1);
  int b  = ml >> 9;
  float acc = cb[c];
#pragma unroll
  for (int j = 0; j < DCONV; ++j) {
    int tl = l - (DCONV - 1) + j;
    if (tl >= 0)
      acc += xz[((size_t)(b * SEQ + tl)) * (2 * DINNER) + c] * cw[c * DCONV + j];
  }
  float s = acc / (1.0f + __expf(-acc));              // silu
  u[idx]   = s;
  u_h[idx] = (_Float16)s;
}

// ---------------- delta = softplus(x_dbl[:, :8] @ w_dt^T + b_dt) ----------------
__global__ void k_delta(const float* __restrict__ xdbl, const float* __restrict__ w_dt,
                        const float* __restrict__ b_dt, float* __restrict__ delta) {
  int idx = blockIdx.x * blockDim.x + threadIdx.x;    // [0, MTOK*DINNER)
  int c = idx & (DINNER - 1);
  int m = idx >> 8;
  float acc = b_dt[c];
  const float* xr = xdbl + (size_t)m * XPROJ_N;
  const float* wr = w_dt + (size_t)c * DTRANK;
#pragma unroll
  for (int r = 0; r < DTRANK; ++r) acc += xr[r] * wr[r];
  delta[idx] = (acc > 20.0f) ? acc : log1pf(__expf(acc));
}

// ---------------- selective scan (sequential over L) + gating epilogue ----------------
// grid (DINNER/64, BATCH), block 64: thread = one (batch, channel); state in registers.
__global__ __launch_bounds__(64)
void k_scan(const float* __restrict__ delta, const float* __restrict__ u,
            const float* __restrict__ xz, const float* __restrict__ xdbl,
            const float* __restrict__ A_log, const float* __restrict__ Dp,
            _Float16* __restrict__ gated) {
  int b = blockIdx.y;
  int c = blockIdx.x * 64 + threadIdx.x;
  float Arow[DSTATE];
#pragma unroll
  for (int s = 0; s < DSTATE; ++s) Arow[s] = -__expf(A_log[(size_t)c * DSTATE + s]);
  const float Dc = Dp[c];
  float hst[DSTATE];
#pragma unroll
  for (int s = 0; s < DSTATE; ++s) hst[s] = 0.0f;

  __shared__ float sBC[2 * DSTATE];
  for (int t = 0; t < SEQ; ++t) {
    int m = b * SEQ + t;
    if (threadIdx.x < 2 * DSTATE)
      sBC[threadIdx.x] = xdbl[(size_t)m * XPROJ_N + DTRANK + threadIdx.x];
    __syncthreads();
    float dlt = delta[(size_t)m * DINNER + c];
    float uu  = u[(size_t)m * DINNER + c];
    float y = 0.0f;
#pragma unroll
    for (int s = 0; s < DSTATE; ++s) {
      hst[s] = __expf(dlt * Arow[s]) * hst[s] + (dlt * sBC[s]) * uu;
      y += hst[s] * sBC[DSTATE + s];
    }
    y += uu * Dc;
    float r = xz[(size_t)m * (2 * DINNER) + DINNER + c];
    float g = y * (r / (1.0f + __expf(-r)));          // y * silu(res)
    gated[(size_t)m * DINNER + c] = (_Float16)g;
    __syncthreads();
  }
}

// ---------------- final rmsnorm (last token) + classifier ----------------
__global__ void k_final(const float* __restrict__ h, const float* __restrict__ normf_w,
                        const float* __restrict__ w_cls, const float* __restrict__ b_cls,
                        float* __restrict__ out) {
  int b = blockIdx.x;              // 32 blocks, 128 threads
  int t = threadIdx.x;
  size_t m = (size_t)b * SEQ + (SEQ - 1);
  float v = h[m * DMODEL + t];
  __shared__ float red[DMODEL];
  red[t] = v * v;
  __syncthreads();
#pragma unroll
  for (int s = DMODEL / 2; s > 0; s >>= 1) {
    if (t < s) red[t] += red[t + s];
    __syncthreads();
  }
  float hn = v * rsqrtf(red[0] / (float)DMODEL + 1e-5f) * normf_w[t];
  __shared__ float p0[DMODEL], p1[DMODEL];
  p0[t] = hn * w_cls[t];
  p1[t] = hn * w_cls[DMODEL + t];
  __syncthreads();
#pragma unroll
  for (int s = DMODEL / 2; s > 0; s >>= 1) {
    if (t < s) { p0[t] += p0[t + s]; p1[t] += p1[t + s]; }
    __syncthreads();
  }
  if (t == 0) {
    out[b * NCLS + 0] = p0[0] + b_cls[0];
    out[b * NCLS + 1] = p1[0] + b_cls[1];
  }
}

// ---------------- host orchestration ----------------
extern "C" void kernel_launch(void* const* d_in, const int* in_sizes, int n_in,
                              void* d_out, int out_size, void* d_ws, size_t ws_size,
                              hipStream_t stream) {
  (void)in_sizes; (void)n_in; (void)out_size; (void)ws_size;
  const float* x        = (const float*)d_in[0];
  const float* w_in     = (const float*)d_in[1];
  const float* b_in     = (const float*)d_in[2];
  const float* w_inproj = (const float*)d_in[3];
  const float* conv_w   = (const float*)d_in[4];
  const float* conv_b   = (const float*)d_in[5];
  const float* w_xproj  = (const float*)d_in[6];
  const float* w_dt     = (const float*)d_in[7];
  const float* b_dt     = (const float*)d_in[8];
  const float* A_log    = (const float*)d_in[9];
  const float* D_param  = (const float*)d_in[10];
  const float* w_out    = (const float*)d_in[11];
  const float* norm_w   = (const float*)d_in[12];
  const float* normf_w  = (const float*)d_in[13];
  const float* w_cls    = (const float*)d_in[14];
  const float* b_cls    = (const float*)d_in[15];
  float* out = (float*)d_out;

  // workspace carve-up (256B aligned)
  char* ws = (char*)d_ws;
  size_t off = 0;
  auto carve = [&](size_t bytes) {
    size_t o = off;
    off = (off + bytes + 255) & ~(size_t)255;
    return (void*)(ws + o);
  };
  _Float16* wip_h  = (_Float16*)carve((size_t)NLAYER * 2 * DINNER * DMODEL * 2);
  _Float16* wxp_h  = (_Float16*)carve((size_t)NLAYER * XPROJ_N * DINNER * 2);
  _Float16* wout_h = (_Float16*)carve((size_t)NLAYER * DMODEL * DINNER * 2);
  float*    hbuf   = (float*)   carve((size_t)MTOK * DMODEL * 4);
  _Float16* xn_h   = (_Float16*)carve((size_t)MTOK * DMODEL * 2);
  float*    xz     = (float*)   carve((size_t)MTOK * 2 * DINNER * 4);
  float*    ubuf   = (float*)   carve((size_t)MTOK * DINNER * 4);
  _Float16* u_h    = (_Float16*)carve((size_t)MTOK * DINNER * 2);
  float*    xdbl   = (float*)   carve((size_t)MTOK * XPROJ_N * 4);
  float*    dbuf   = (float*)   carve((size_t)MTOK * DINNER * 4);
  _Float16* gate_h = (_Float16*)carve((size_t)MTOK * DINNER * 2);

  // weights -> f16 (layouts already [N,K] row-major as the GEMM expects)
  {
    int n1 = NLAYER * 2 * DINNER * DMODEL;
    int n2 = NLAYER * XPROJ_N * DINNER;
    int n3 = NLAYER * DMODEL * DINNER;
    k_cast_f16<<<(n1 + 255) / 256, 256, 0, stream>>>(w_inproj, wip_h, n1);
    k_cast_f16<<<(n2 + 255) / 256, 256, 0, stream>>>(w_xproj,  wxp_h, n2);
    k_cast_f16<<<(n3 + 255) / 256, 256, 0, stream>>>(w_out,    wout_h, n3);
  }

  // h = x @ w_in^T + b_in
  k_in_proj<<<(MTOK * DMODEL) / 256, 256, 0, stream>>>(x, w_in, b_in, hbuf);

  for (int i = 0; i < NLAYER; ++i) {
    const _Float16* wip_l  = wip_h  + (size_t)i * 2 * DINNER * DMODEL;
    const _Float16* wxp_l  = wxp_h  + (size_t)i * XPROJ_N * DINNER;
    const _Float16* wout_l = wout_h + (size_t)i * DMODEL * DINNER;
    const float* cw_l  = conv_w  + (size_t)i * DINNER * DCONV;
    const float* cb_l  = conv_b  + (size_t)i * DINNER;
    const float* wdt_l = w_dt    + (size_t)i * DINNER * DTRANK;
    const float* bdt_l = b_dt    + (size_t)i * DINNER;
    const float* Al_l  = A_log   + (size_t)i * DINNER * DSTATE;
    const float* Dp_l  = D_param + (size_t)i * DINNER;
    const float* nw_l  = norm_w  + (size_t)i * DMODEL;

    // xn = f16(rmsnorm(h))
    k_rmsnorm_f16<<<MTOK, DMODEL, 0, stream>>>(hbuf, nw_l, xn_h);
    // xz = xn @ w_inproj^T   [16384 x 512], K=128
    wmma_gemm<DMODEL, 2 * DINNER, false>
        <<<dim3(MTOK / 128, (2 * DINNER) / 64), 256, 0, stream>>>(xn_h, wip_l, xz, nullptr);
    // u = silu(causal depthwise conv(xz[:, :256]))
    k_conv_silu<<<(MTOK * DINNER) / 256, 256, 0, stream>>>(xz, cw_l, cb_l, ubuf, u_h);
    // x_dbl = u @ w_xproj^T  [16384 x 40], K=256 (N padded to one 64-tile)
    wmma_gemm<DINNER, XPROJ_N, false>
        <<<dim3(MTOK / 128, 1), 256, 0, stream>>>(u_h, wxp_l, xdbl, nullptr);
    // delta = softplus(x_dbl[:, :8] @ w_dt^T + b_dt)
    k_delta<<<(MTOK * DINNER) / 256, 256, 0, stream>>>(xdbl, wdt_l, bdt_l, dbuf);
    // selective scan + (+u*D) + gating with silu(res) -> gated (f16)
    k_scan<<<dim3(DINNER / 64, BATCH), 64, 0, stream>>>(dbuf, ubuf, xz, xdbl, Al_l, Dp_l, gate_h);
    // h = gated @ w_out^T + h   [16384 x 128], K=256, fused residual
    wmma_gemm<DINNER, DMODEL, true>
        <<<dim3(MTOK / 128, DMODEL / 64), 256, 0, stream>>>(gate_h, wout_l, hbuf, hbuf);
  }

  // final rmsnorm (last token only) + classifier
  k_final<<<BATCH, DMODEL, 0, stream>>>(hbuf, normf_w, w_cls, b_cls, out);
}
